// PruningQuantizationWrapper_6923487282600
// MI455X (gfx1250) — matrix operand order repaired and checked by reading, not compiled, not run on hardware
//
#include <hip/hip_runtime.h>
#include <hip/hip_bf16.h>

typedef __attribute__((ext_vector_type(16))) _Float16 v16h;
typedef __attribute__((ext_vector_type(8)))  _Float16 v8h;
typedef __attribute__((ext_vector_type(8)))  float    v8f;
typedef __attribute__((ext_vector_type(4)))  int      v4i;

#define COUT 512
#define CIN  512
#define KDIM 4608          // CIN * 9
#define HW   1024          // 32*32
#define NPIX 32768         // 32 * HW
#define PADW 34            // padded image width/height
#define PSTR (PADW*PADW)   // 1156 halves per padded channel image
#define BM   128
#define BN   128
#define KB   32

// ---- CDNA5 async global->LDS (guarded; falls back to register copy) ----
#if defined(__has_builtin)
#if __has_builtin(__builtin_amdgcn_global_load_async_to_lds_b128)
#define USE_ASYNC_LDS 1
#endif
#if __has_builtin(__builtin_amdgcn_s_wait_asynccnt)
#define HAVE_WAIT_ASYNC 1
#endif
#endif

typedef __attribute__((address_space(1))) v4i* gv4i_p;   // global int4*
typedef __attribute__((address_space(3))) v4i* lv4i_p;   // LDS int4*

__device__ __forceinline__ void copy_b128_to_lds(const _Float16* g, _Float16* l) {
#if defined(USE_ASYNC_LDS)
    __builtin_amdgcn_global_load_async_to_lds_b128((gv4i_p)g, (lv4i_p)l, 0, 0);
#else
    *reinterpret_cast<v8h*>(l) = *reinterpret_cast<const v8h*>(g);
#endif
}

__device__ __forceinline__ void wait_async_copies() {
#if defined(USE_ASYNC_LDS)
#if defined(HAVE_WAIT_ASYNC)
    __builtin_amdgcn_s_wait_asynccnt(0);
#else
    asm volatile("s_wait_asynccnt 0x0" ::: "memory");
#endif
#endif
}

__device__ __forceinline__ float sigmoidf_(float x) { return 1.0f / (1.0f + __expf(-x)); }

// ---------------------------------------------------------------------------
// 1) global max |w * gate|
// ---------------------------------------------------------------------------
__global__ void pq_absmax_kernel(const float* __restrict__ w,
                                 const float* __restrict__ alpha,
                                 unsigned int* __restrict__ wmax_bits) {
    __shared__ float red[256];
    float m = 0.0f;
    const int total = COUT * KDIM;
    for (int i = blockIdx.x * blockDim.x + threadIdx.x; i < total;
         i += gridDim.x * blockDim.x) {
        const int o = i / KDIM;
        m = fmaxf(m, fabsf(w[i]) * sigmoidf_(alpha[o]));
    }
    red[threadIdx.x] = m;
    __syncthreads();
    for (int s = 128; s > 0; s >>= 1) {
        if (threadIdx.x < s) red[threadIdx.x] = fmaxf(red[threadIdx.x], red[threadIdx.x + s]);
        __syncthreads();
    }
    if (threadIdx.x == 0)
        atomicMax(wmax_bits, __float_as_uint(red[0]));  // non-negative: bit order == value order
}

// ---------------------------------------------------------------------------
// 2) w_q = sum_k pi_k * Q_{b_k}(w * gate)  ->  f16 GEMM A matrix [COUT, KDIM]
// ---------------------------------------------------------------------------
__global__ void pq_build_wq_kernel(const float* __restrict__ w,
                                   const float* __restrict__ alpha,
                                   const float* __restrict__ beta,
                                   const float* __restrict__ gumbel,
                                   const unsigned int* __restrict__ wmax_bits,
                                   _Float16* __restrict__ wh) {
    float l0 = beta[0] + gumbel[0], l1 = beta[1] + gumbel[1];
    float l2 = beta[2] + gumbel[2], l3 = beta[3] + gumbel[3];
    float lm = fmaxf(fmaxf(l0, l1), fmaxf(l2, l3));
    float e0 = __expf(l0 - lm), e1 = __expf(l1 - lm);
    float e2 = __expf(l2 - lm), e3 = __expf(l3 - lm);
    float inv = 1.0f / (e0 + e1 + e2 + e3);
    const float pi[4] = {e0 * inv, e1 * inv, e2 * inv, e3 * inv};
    const float nv[4] = {1.0f, 7.0f, 127.0f, 32767.0f};  // 2^(b-1)-1, b=2,4,8,16

    const float wmax = fmaxf(__uint_as_float(*wmax_bits), 1e-8f);

    const int total = COUT * KDIM;
    for (int i = blockIdx.x * blockDim.x + threadIdx.x; i < total;
         i += gridDim.x * blockDim.x) {
        const int o = i / KDIM;
        const float g = w[i] * sigmoidf_(alpha[o]);
        float acc = 0.0f;
#pragma unroll
        for (int k = 0; k < 4; ++k) {
            const float s = wmax / nv[k];
            float q = rintf(g / s);                       // round-half-even == jnp.round
            q = fminf(fmaxf(q, -nv[k] - 1.0f), nv[k]);
            acc += pi[k] * q * s;
        }
        wh[i] = (_Float16)acc;
    }
}

// ---------------------------------------------------------------------------
// 3) activations: fp32 NCHW -> f16 zero-padded [32,512,34,34]
// ---------------------------------------------------------------------------
__global__ void pq_pad_x_kernel(const float* __restrict__ x,
                                _Float16* __restrict__ xp, int total) {
    const int i = blockIdx.x * blockDim.x + threadIdx.x;
    if (i >= total) return;
    const int iw = i % PADW;
    const int t  = i / PADW;
    const int ih = t % PADW;
    const int nc = t / PADW;            // fused (n, c) index
    float v = 0.0f;
    if (iw >= 1 && iw <= 32 && ih >= 1 && ih <= 32)
        v = x[((size_t)nc * 32 + (ih - 1)) * 32 + (iw - 1)];
    xp[i] = (_Float16)v;
}

// ---------------------------------------------------------------------------
// helpers for the pipelined conv
// ---------------------------------------------------------------------------
__device__ __forceinline__ void stage_A_async(const _Float16* __restrict__ wh,
                                              _Float16* asb, int block_m, int kb, int tid) {
#pragma unroll
    for (int c = tid; c < (BM * KB) / 8; c += 256) {
        const int m    = c >> 2;
        const int kofs = (c & 3) * 8;
        const size_t gidx = (size_t)(block_m + m) * KDIM + (kb + kofs);
        copy_b128_to_lds(wh + gidx, &asb[m * KB + kofs]);
        __builtin_prefetch(wh + gidx + 2 * KB, 0, 1);   // global_prefetch_b8: 2 slices ahead
    }
}

__device__ __forceinline__ const _Float16* bsrc_ptr(const _Float16* __restrict__ xp,
                                                    int k, int ni, int h0, int w0) {
    const int ci = k / 9;
    const int r  = k - ci * 9;
    const int kh = r / 3;
    const int kw = r - kh * 3;
    return xp + ((size_t)(ni * CIN + ci) * PADW + (h0 + kh)) * PADW + (w0 + kw);
}

// ---------------------------------------------------------------------------
// 4) implicit-GEMM conv, v_wmma_f32_16x16x32_f16, double-buffered async pipeline
//    block tile 128(M=Cout) x 128(N=pixels), K-step 32
//    8 waves: (wm 0..3 -> 32 M rows) x (wn 0..1 -> 64 N cols); 8 WMMAs/wave/step
// ---------------------------------------------------------------------------
__global__ __launch_bounds__(256) void pq_conv_wmma_kernel(
    const _Float16* __restrict__ xp,   // [32, 512, 34, 34] f16 padded
    const _Float16* __restrict__ wh,   // [512, 4608] f16 row-major
    const float* __restrict__ bias,    // [512]
    float* __restrict__ out) {         // [32, 512, 32, 32] f32

    __shared__ __align__(16) _Float16 As[2][BM * KB];   // [m][k]
    __shared__ __align__(16) _Float16 Bs[2][BN * KB];   // [p][k]

    const int tid  = threadIdx.x;
    const int lane = tid & 31;
    const int wave = tid >> 5;
    const int wm   = wave & 3;
    const int wn   = wave >> 2;
    const int block_p = blockIdx.x * BN;
    const int block_m = blockIdx.y * BM;

    v8f acc[2][4] = {};

    // B staging: thread -> (one k, 16 contiguous pixels); segment stays in one image row
    const int bk  = tid >> 3;
    const int bp0 = (tid & 7) * 16;
    const int p0 = block_p + bp0;
    const int ni = p0 >> 10;
    const int h0 = (p0 & 1023) >> 5;
    const int w0 = p0 & 31;

    _Float16 breg[16];

    // ---- prologue: fill buffer 0 ----
    stage_A_async(wh, As[0], block_m, 0, tid);
    {
        const _Float16* src = bsrc_ptr(xp, bk, ni, h0, w0);
#pragma unroll
        for (int j = 0; j < 16; ++j) Bs[0][(bp0 + j) * KB + bk] = src[j];
    }
    wait_async_copies();
    __syncthreads();

    for (int kb = 0; kb < KDIM; kb += KB) {
        const int  cur  = (kb >> 5) & 1;
        const int  nxt  = cur ^ 1;
        const bool more = (kb + KB) < KDIM;

        // ---- issue next tile: async A -> LDS, global B -> regs (overlaps WMMAs) ----
        if (more) {
            stage_A_async(wh, As[nxt], block_m, kb + KB, tid);
            const _Float16* src = bsrc_ptr(xp, kb + KB + bk, ni, h0, w0);
#pragma unroll
            for (int j = 0; j < 16; ++j) breg[j] = src[j];
        }

        // ---- compute on current buffer ----
        v16h af[2];
#pragma unroll
        for (int s = 0; s < 2; ++s) {
            const int row = wm * 32 + s * 16 + (lane & 15);
            const int k0  = (lane >> 4) * 8;
            const v8h lo = *reinterpret_cast<const v8h*>(&As[cur][row * KB + k0]);
            const v8h hi = *reinterpret_cast<const v8h*>(&As[cur][row * KB + k0 + 16]);
            af[s] = __builtin_shufflevector(lo, hi, 0,1,2,3,4,5,6,7,8,9,10,11,12,13,14,15);
        }
        const int bkk = (lane >> 4) * 8;
        const int pl  = wn * 64 + (lane & 15);
#pragma unroll
        for (int f = 0; f < 4; ++f) {
            const int col = (pl + f * 16) * KB + bkk;
            const v8h lo = *reinterpret_cast<const v8h*>(&Bs[cur][col]);
            const v8h hi = *reinterpret_cast<const v8h*>(&Bs[cur][col + 16]);
            const v16h bf =
                __builtin_shufflevector(lo, hi, 0,1,2,3,4,5,6,7,8,9,10,11,12,13,14,15);
            acc[0][f] = __builtin_amdgcn_wmma_f32_16x16x32_f16(
                false, af[0], false, bf, (short)0, acc[0][f], false, false);
            acc[1][f] = __builtin_amdgcn_wmma_f32_16x16x32_f16(
                false, af[1], false, bf, (short)0, acc[1][f], false, false);
        }

        // ---- flush staged B regs to next buffer, then close the step ----
        if (more) {
#pragma unroll
            for (int j = 0; j < 16; ++j) Bs[nxt][(bp0 + j) * KB + bk] = breg[j];
        }
        wait_async_copies();
        __syncthreads();
    }

    // ---- epilogue: C layout lane = {m/8 half, n}, vgpr j = m%8 ----
    const int nn = lane & 15;
    const int mo = (lane >> 4) * 8;
#pragma unroll
    for (int f = 0; f < 4; ++f) {
        const int p   = block_p + wn * 64 + f * 16 + nn;
        const int pni = p >> 10;
        const int phw = p & 1023;
        float* op = out + (size_t)pni * COUT * HW + phw;
#pragma unroll
        for (int s = 0; s < 2; ++s)
#pragma unroll
            for (int j = 0; j < 8; ++j) {
                const int o = block_m + wm * 32 + s * 16 + mo + j;
                op[(size_t)o * HW] = acc[s][f][j] + bias[o];
            }
    }
}

// ---------------------------------------------------------------------------
extern "C" void kernel_launch(void* const* d_in, const int* in_sizes, int n_in,
                              void* d_out, int out_size, void* d_ws, size_t ws_size,
                              hipStream_t stream) {
    const float* x      = (const float*)d_in[0];
    const float* weight = (const float*)d_in[1];
    const float* bias   = (const float*)d_in[2];
    const float* alpha  = (const float*)d_in[3];
    const float* beta   = (const float*)d_in[4];
    const float* gumbel = (const float*)d_in[5];
    float* out = (float*)d_out;

    char* ws = (char*)d_ws;
    const size_t xp_bytes = (size_t)32 * CIN * PSTR * sizeof(_Float16);
    _Float16* xp = (_Float16*)ws;
    _Float16* wh = (_Float16*)(ws + xp_bytes);
    unsigned int* wmax = (unsigned int*)(ws + xp_bytes + (size_t)COUT * KDIM * sizeof(_Float16));

    (void)hipMemsetAsync(wmax, 0, sizeof(unsigned int), stream);

    pq_absmax_kernel<<<512, 256, 0, stream>>>(weight, alpha, wmax);

    pq_build_wq_kernel<<<(COUT * KDIM) / 256, 256, 0, stream>>>(
        weight, alpha, beta, gumbel, wmax, wh);

    const int pad_total = 32 * CIN * PSTR;
    pq_pad_x_kernel<<<(pad_total + 255) / 256, 256, 0, stream>>>(x, xp, pad_total);

    dim3 grid(NPIX / BN, COUT / BM);
    pq_conv_wmma_kernel<<<grid, 256, 0, stream>>>(xp, wh, bias, out);
}
